// _BaseMemoryBranch_6193342841004
// MI455X (gfx1250) — matrix operand order repaired and checked by reading, not compiled, not run on hardware
//
#include <hip/hip_runtime.h>
#include <hip/hip_bf16.h>

// Fused memory-attention: out = softmax( unfold(x) @ mem^T * t/24 ) @ mem
// Shapes: x[16,64,64,64] f32, mem[512,576] f32, out[16,4096,576] f32.
// Strategy (MI455X): output-BW bound (~151MB out vs ~78 GFLOP). Fuse everything
// per 16-query tile so the 134MB sim matrix never touches HBM. bf16 WMMA
// (v_wmma_f32_16x16x32_bf16) with f32 accumulate; mem pre-converted to bf16
// (row-major + transposed) in 1.2MB of workspace that stays L2-resident.

typedef __attribute__((ext_vector_type(16))) __bf16 v16bf;
typedef __attribute__((ext_vector_type(8)))  __bf16 v8bf;
typedef __attribute__((ext_vector_type(8)))  float  v8f;

union V16U { v16bf v; v8bf h[2]; };

__device__ __forceinline__ v16bf load_bf16_frag(const __bf16* p0, const __bf16* p1) {
  V16U u;
  u.h[0] = *(const v8bf*)p0;   // 16B load
  u.h[1] = *(const v8bf*)p1;   // 16B load
  return u.v;
}

// ---------------- prologue: f32 memory -> bf16 row-major + transposed ----------------
__global__ __launch_bounds__(256) void convert_mem(const float* __restrict__ mem,
                                                   __bf16* __restrict__ mem16,   // [512][576]
                                                   __bf16* __restrict__ memT16)  // [576][512]
{
  int i = blockIdx.x * blockDim.x + threadIdx.x;
  if (i < 512 * 576) {
    float v = mem[i];
    __bf16 b = (__bf16)v;
    int m = i / 576;
    int d = i - m * 576;
    mem16[i] = b;
    memT16[d * 512 + m] = b;
  }
}

// ---------------- fused attention kernel ----------------
// LDS map (bytes):
//   [0,      18944)  xa  : A-tile bf16 [16][592]  (padded stride)
//   [0,      16384)  att : bf16 [16][512]         (overlays xa after GEMM1)
//   [18944,  51712)  sim : f32  [16][512]
//   [51712,  52736)  red : f32  [16][16]
//   [52736,  52800)  inv : f32  [16]
#define SM_SIM  18944
#define SM_RED  51712
#define SM_INV  52736
#define SM_SIZE 52800
#define XA_STR  592

__global__ __launch_bounds__(256) void attn_kernel(const float* __restrict__ x,
                                                   const __bf16* __restrict__ mem16,
                                                   const __bf16* __restrict__ memT16,
                                                   const float* __restrict__ temp,
                                                   float* __restrict__ out)
{
  __shared__ __align__(16) unsigned char smem[SM_SIZE];
  __bf16* xa  = (__bf16*)(smem);
  __bf16* att = (__bf16*)(smem);            // overlay (xa dead after GEMM1)
  float*  sim = (float*)(smem + SM_SIM);
  float*  red = (float*)(smem + SM_RED);
  float*  inv = (float*)(smem + SM_INV);

  const int tid  = threadIdx.x;
  const int lane = tid & 31;
  const int wave = tid >> 5;
  const int half = lane >> 4;     // which 16-lane half of the wave
  const int l16  = lane & 15;

  const int t   = blockIdx.x;     // 16-query tile id, 0..4095
  const int n0  = t << 4;         // global query base
  const int b   = n0 >> 12;       // image index
  const int nin = n0 & 4095;
  const int h   = nin >> 6;       // all 16 queries share this row
  const int w0  = nin & 63;

  // ---- phase 1: unfold 3x3x64 patches for 16 queries into LDS (bf16) ----
  for (int idx = tid; idx < 16 * 576; idx += 256) {
    int row = idx / 576;
    int d   = idx - row * 576;
    int c   = d / 9;
    int k   = d - c * 9;
    int hh  = h + (k / 3) - 1;
    int ww  = w0 + row + (k % 3) - 1;
    float v = 0.0f;
    if (hh >= 0 && hh < 64 && ww >= 0 && ww < 64)
      v = x[((((b << 6) + c) << 6) + hh) * 64 + ww];
    xa[row * XA_STR + d] = (__bf16)v;
  }
  __syncthreads();

  const float alpha = temp[0] * (1.0f / 24.0f);   // temperature / sqrt(576)

  // WMMA bf16 operand lane offsets (ISA 16-bit A/B layouts):
  const int kA0 = half * 8;        // A: lanes<16 hold K 0-7 & 16-23; lanes>=16 hold 8-15 & 24-31
  const int kA1 = 16 + half * 8;
  const int kB  = half * 16;       // B: same K layout as A (kA0/kA1)

  __builtin_prefetch(mem16 + (size_t)(wave * 64 + l16) * 576, 0, 0);

  // ---- phase 2: sim[16,512] = Xa[16,576] @ mem^T, 4 N-tiles per wave ----
  for (int nt = wave; nt < 32; nt += 8) {
    const int nbase = nt << 4;
    const __bf16* brow = mem16 + (size_t)(nbase + l16) * 576;  // memory row = B column
    const __bf16* arow = xa + l16 * XA_STR;
    v8f acc = {};
    for (int kc = 0; kc < 18; ++kc) {
      const int kb = kc * 32;
      v16bf A  = load_bf16_frag(arow + kb + kA0, arow + kb + kA1);
      v16bf Bf = load_bf16_frag(brow + kb + kA0, brow + kb + kA1);
      acc = __builtin_amdgcn_wmma_f32_16x16x32_bf16(false, A, false, Bf,
                                                    (short)0, acc, false, false);
    }
#pragma unroll
    for (int r = 0; r < 8; ++r)
      sim[(r + half * 8) * 512 + nbase + l16] = acc[r] * alpha;
  }
  __syncthreads();

  // ---- phase 3: row softmax over 512 (16 threads per row) ----
  {
    const int row = tid >> 4;
    const int sl  = tid & 15;
    float mx = -3.4e38f;
    for (int j = sl; j < 512; j += 16) mx = fmaxf(mx, sim[row * 512 + j]);
    red[row * 16 + sl] = mx;
    __syncthreads();
    if (sl == 0) {
      float m2 = red[row * 16];
      for (int q = 1; q < 16; ++q) m2 = fmaxf(m2, red[row * 16 + q]);
      red[row * 16] = m2;
    }
    __syncthreads();
    const float rmax = red[row * 16];
    __syncthreads();                      // red reads done before reuse for sums
    float s = 0.0f;
    for (int j = sl; j < 512; j += 16) {
      float e = __expf(sim[row * 512 + j] - rmax);
      att[row * 512 + j] = (__bf16)e;     // unnormalized exp, bf16 (<= 1.0)
      s += e;
    }
    red[row * 16 + sl] = s;
    __syncthreads();
    if (sl == 0) {
      float tot = 0.0f;
      for (int q = 0; q < 16; ++q) tot += red[row * 16 + q];
      inv[row] = 1.0f / tot;
    }
  }
  __syncthreads();

  float invr[8];
#pragma unroll
  for (int r = 0; r < 8; ++r) invr[r] = inv[r + half * 8];

  // ---- phase 4: out[16,576] = att[16,512] @ mem, 4-5 D-tiles per wave ----
  for (int ot = wave; ot < 36; ot += 8) {
    const int dbase = ot << 4;
    const __bf16* brow = memT16 + (size_t)(dbase + l16) * 512; // feature column of mem
    const __bf16* arow = att + l16 * 512;
    v8f acc = {};
    for (int kc = 0; kc < 16; ++kc) {
      const int kb = kc * 32;
      v16bf A  = load_bf16_frag(arow + kb + kA0, arow + kb + kA1);
      v16bf Bf = load_bf16_frag(brow + kb + kA0, brow + kb + kA1);
      acc = __builtin_amdgcn_wmma_f32_16x16x32_bf16(false, A, false, Bf,
                                                    (short)0, acc, false, false);
    }
#pragma unroll
    for (int r = 0; r < 8; ++r)
      out[(size_t)(n0 + r + half * 8) * 576 + dbase + l16] = acc[r] * invr[r];
  }
}

extern "C" void kernel_launch(void* const* d_in, const int* in_sizes, int n_in,
                              void* d_out, int out_size, void* d_ws, size_t ws_size,
                              hipStream_t stream) {
  const float* x    = (const float*)d_in[0];   // [16,64,64,64]
  const float* mem  = (const float*)d_in[1];   // [512,576]
  const float* temp = (const float*)d_in[2];   // [1]
  float* out = (float*)d_out;                  // [16,4096,576]

  // workspace: bf16 memory, row-major then transposed (589824 B each)
  __bf16* mem16  = (__bf16*)d_ws;
  __bf16* memT16 = (__bf16*)((char*)d_ws + 512 * 576 * 2);

  convert_mem<<<(512 * 576 + 255) / 256, 256, 0, stream>>>(mem, mem16, memT16);
  attn_kernel<<<4096, 256, 0, stream>>>(x, mem16, memT16, temp, out);

  (void)in_sizes; (void)n_in; (void)out_size; (void)ws_size;
}
